// VarianceAdaptor_8323646620046
// MI455X (gfx1250) — matrix operand order repaired and checked by reading, not compiled, not run on hardware
//
#include <hip/hip_runtime.h>

typedef __attribute__((ext_vector_type(16))) __bf16 v16bf;
typedef __attribute__((ext_vector_type(8)))  float  v8f;

#define BB     32
#define SS     512
#define HH     256
#define FF     256
#define BS     16384      // BB*SS
#define KDIM   768        // 3*HH
#define KC_N   24         // KDIM/32
#define CT_N   16         // FF/16
#define RT_N   1024       // BS/16
#define MAXMEL 2048
#define NBIN1  255

// ---------------------------------------------------------------------------
// Fragment index maps (CDNA5 ISA 7.12.2, 16-bit A 16x32 / B 32x16, wave32)
// ---------------------------------------------------------------------------
__device__ __forceinline__ int a_k_of(int lane, int e) {
    int hi = lane >> 4;
    int v  = e >> 1;
    int base = (v < 4) ? (2 * v + 8 * hi) : (16 + 2 * (v - 4) + 8 * hi);
    return base + (e & 1);
}
__device__ __forceinline__ int b_k_of(int lane, int e) {
    return a_k_of(lane, e);
}

// ---------------------------------------------------------------------------
// im2col (K=3, SAME, per-batch zero pad) -> A-fragment-ordered bf16 buffer
// layout: [rt=1024][kc=24][lane=32][e=16]
// ---------------------------------------------------------------------------
template <typename T>
__global__ void im2col_frag_kernel(const T* __restrict__ src, __bf16* __restrict__ dst) {
    long gid = (long)blockIdx.x * 256 + threadIdx.x;   // total 1024*24*512 = 12582912
    if (gid >= (long)RT_N * KC_N * 512) return;
    int e    = (int)(gid & 15);
    int lane = (int)((gid >> 4) & 31);
    long rk  = gid >> 9;              // rt*24 + kc
    int kc   = (int)(rk % KC_N);
    int rt   = (int)(rk / KC_N);
    int row  = rt * 16 + (lane & 15); // global row in [0, BS)
    int K    = kc * 32 + a_k_of(lane, e);
    int kk   = K >> 8;                // 0..2 (conv tap)
    int c    = K & 255;               // channel
    int s    = row & (SS - 1);
    int b    = row >> 9;
    int ss   = s + kk - 1;
    float val = (ss < 0 || ss >= SS) ? 0.0f
                                     : (float)src[((long)(b * SS + ss)) * HH + c];
    dst[gid] = (__bf16)val;
}

// ---------------------------------------------------------------------------
// Weight [768][256] f32 -> B-fragment-ordered bf16: [kc=24][ct=16][lane=32][e=16]
// ---------------------------------------------------------------------------
__global__ void wfrag_kernel(const float* __restrict__ W, __bf16* __restrict__ dst) {
    int gid = blockIdx.x * 256 + threadIdx.x;          // total 24*16*512 = 196608
    if (gid >= KC_N * CT_N * 512) return;
    int e    = gid & 15;
    int lane = (gid >> 4) & 31;
    int ct   = (gid >> 9) & 15;
    int kc   = gid >> 13;
    int K    = kc * 32 + b_k_of(lane, e);
    int N    = ct * 16 + (lane & 15);
    dst[gid] = (__bf16)W[(long)K * FF + N];
}

// ---------------------------------------------------------------------------
// Fused GEMM (im2col conv) + bias + ReLU + LayerNorm.
// MODE 0: write LN output as bf16 [BS][256]
// MODE 1: additionally dot with wl + bl, mask, scale -> f32 [BS]
// Block: 128 threads (4 waves), one 16-row x 256-col output tile per block.
// K loop fully unrolled: static load offsets, fixed accumulator VGPRs, loads
// hoistable across WMMAs for latency hiding.
// ---------------------------------------------------------------------------
template <int MODE>
__global__ __launch_bounds__(128)
void gemm_ln_kernel(const __bf16* __restrict__ Af, const __bf16* __restrict__ Bf,
                    const float* __restrict__ bias, const float* __restrict__ gamma,
                    const float* __restrict__ beta,
                    __bf16* __restrict__ Hout,
                    float* __restrict__ Dout, const float* __restrict__ wl,
                    const float* __restrict__ bl, const unsigned char* __restrict__ mask,
                    const float* __restrict__ ctrl) {
    __shared__ float tile[16][FF + 8];
    __shared__ float red[16][8][2];
    __shared__ float mv[16][2];

    int rt   = blockIdx.x;          // row tile (16 rows)
    int tid  = threadIdx.x;
    int wave = tid >> 5;
    int lane = tid & 31;
    int n_lo = lane & 15;
    int hi   = lane >> 4;

    // per-lane fragment base pointers
    const __bf16* Aw = Af + (long)rt * KC_N * 512 + lane * 16;
    const __bf16* Bw = Bf + (long)(wave * 4) * 512 + lane * 16;

    v8f acc0 = {}, acc1 = {}, acc2 = {}, acc3 = {};
#pragma unroll
    for (int kc = 0; kc < KC_N; ++kc) {
        if (kc + 2 < KC_N)
            __builtin_prefetch(Aw + (long)(kc + 2) * 512, 0, 1);
        v16bf a  = *(const v16bf*)(Aw + (long)kc * 512);
        const __bf16* Bk = Bw + (long)kc * (CT_N * 512);
        v16bf b0 = *(const v16bf*)(Bk + 0 * 512);
        v16bf b1 = *(const v16bf*)(Bk + 1 * 512);
        v16bf b2 = *(const v16bf*)(Bk + 2 * 512);
        v16bf b3 = *(const v16bf*)(Bk + 3 * 512);
        acc0 = __builtin_amdgcn_wmma_f32_16x16x32_bf16(false, a, false, b0, (short)0, acc0, false, false);
        acc1 = __builtin_amdgcn_wmma_f32_16x16x32_bf16(false, a, false, b1, (short)0, acc1, false, false);
        acc2 = __builtin_amdgcn_wmma_f32_16x16x32_bf16(false, a, false, b2, (short)0, acc2, false, false);
        acc3 = __builtin_amdgcn_wmma_f32_16x16x32_bf16(false, a, false, b3, (short)0, acc3, false, false);
    }

    // bias + ReLU -> LDS tile (C layout: VGPR r -> M = r + 8*hi, N = lane&15)
    {
        v8f accs[4] = {acc0, acc1, acc2, acc3};
#pragma unroll
        for (int ct = 0; ct < 4; ++ct) {
            int N = (wave * 4 + ct) * 16 + n_lo;
            float bia = bias[N];
#pragma unroll
            for (int r = 0; r < 8; ++r) {
                float v = accs[ct][r] + bia;
                tile[hi * 8 + r][N] = v > 0.0f ? v : 0.0f;
            }
        }
    }
    __syncthreads();

    // LayerNorm over 256 channels per row: 8 threads / row
    int row = tid >> 3;
    int seg = tid & 7;
    {
        float s = 0.0f, s2 = 0.0f;
#pragma unroll
        for (int j = 0; j < 32; ++j) {
            float v = tile[row][seg * 32 + j];
            s += v; s2 += v * v;
        }
        red[row][seg][0] = s;
        red[row][seg][1] = s2;
    }
    __syncthreads();
    if (tid < 16) {
        float S = 0.0f, S2 = 0.0f;
#pragma unroll
        for (int j = 0; j < 8; ++j) { S += red[tid][j][0]; S2 += red[tid][j][1]; }
        float m   = S * (1.0f / 256.0f);
        float var = S2 * (1.0f / 256.0f) - m * m;
        mv[tid][0] = m;
        mv[tid][1] = rsqrtf(var + 1e-5f);
    }
    __syncthreads();

    float m  = mv[row][0];
    float is = mv[row][1];

    if (MODE == 0) {
        long rowg = (long)(rt * 16 + row) * FF;
#pragma unroll
        for (int j = 0; j < 32; ++j) {
            int n = seg * 32 + j;
            float y = (tile[row][n] - m) * is * gamma[n] + beta[n];
            Hout[rowg + n] = (__bf16)y;
        }
    } else {
        float d = 0.0f;
#pragma unroll
        for (int j = 0; j < 32; ++j) {
            int n = seg * 32 + j;
            float y = (tile[row][n] - m) * is * gamma[n] + beta[n];
            d += y * wl[n];
        }
        red[row][seg][0] = d;
        __syncthreads();
        if (tid < 16) {
            float D = 0.0f;
#pragma unroll
            for (int j = 0; j < 8; ++j) D += red[tid][j][0];
            int grow = rt * 16 + tid;
            float sc = ctrl ? ctrl[0] : 1.0f;
            float o  = (D + bl[0]) * sc;
            Dout[grow] = mask[grow] ? 0.0f : o;
        }
    }
}

// ---------------------------------------------------------------------------
// bucketize (lower_bound over 255 bins) + embedding add: xout = xin + emb[idx]
// ---------------------------------------------------------------------------
__global__ void add_emb_kernel(const float* __restrict__ pred, const float* __restrict__ bins,
                               const float* __restrict__ emb, const float* __restrict__ xin,
                               float* __restrict__ xout) {
    int bs = blockIdx.x;
    int h  = threadIdx.x;
    float v = pred[bs];
    int lo = 0, hi = NBIN1;
    while (lo < hi) {
        int mid = (lo + hi) >> 1;
        if (bins[mid] < v) lo = mid + 1; else hi = mid;
    }
    long off = (long)bs * HH + h;
    xout[off] = xin[off] + emb[(long)lo * HH + h];
}

// ---------------------------------------------------------------------------
// duration: dur = max(rint(exp(log_dur)-1)*dc, 0); cumsum per batch; mel_len
// one block per batch, 512 threads
// ---------------------------------------------------------------------------
__global__ __launch_bounds__(512)
void duration_kernel(const float* __restrict__ log_dur, const float* __restrict__ dctl,
                     float* __restrict__ dur_out, int* __restrict__ cum,
                     float* __restrict__ mel_len_f, int* __restrict__ mel_len_i) {
    __shared__ int sc[SS];
    __shared__ int sc2[SS];
    int b = blockIdx.x;
    int t = threadIdx.x;
    float dc = dctl[0];
    float d  = rintf(expf(log_dur[b * SS + t]) - 1.0f) * dc;
    d = d > 0.0f ? d : 0.0f;
    dur_out[b * SS + t] = d;
    sc[t] = (int)d;
    __syncthreads();
    int* a  = sc;
    int* bb = sc2;
    for (int off = 1; off < SS; off <<= 1) {
        int v = a[t] + ((t >= off) ? a[t - off] : 0);
        bb[t] = v;
        __syncthreads();
        int* tmp = a; a = bb; bb = tmp;
    }
    cum[b * SS + t] = a[t];
    if (t == SS - 1) {
        int ml = a[t] < MAXMEL ? a[t] : MAXMEL;
        mel_len_i[b] = ml;
        mel_len_f[b] = (float)ml;
    }
}

// ---------------------------------------------------------------------------
// length regulator gather: idx = upper_bound(cum_b, t), clip; mask & emit
// one block per (b, t) frame, 256 threads over H
// ---------------------------------------------------------------------------
__global__ void gather_kernel(const float* __restrict__ x3, const int* __restrict__ cum,
                              const int* __restrict__ mel_len, float* __restrict__ xexp,
                              float* __restrict__ mel_mask) {
    int bt = blockIdx.x;
    int b  = bt >> 11;
    int t  = bt & (MAXMEL - 1);
    const int* c = cum + b * SS;
    int lo = 0, hi = SS;
    while (lo < hi) {
        int mid = (lo + hi) >> 1;
        if (c[mid] <= t) lo = mid + 1; else hi = mid;
    }
    int idx = lo < (SS - 1) ? lo : (SS - 1);
    bool msk = (t >= mel_len[b]);
    int h = threadIdx.x;
    xexp[(long)bt * HH + h] = msk ? 0.0f : x3[((long)b * SS + idx) * HH + h];
    if (h == 0) mel_mask[bt] = msk ? 1.0f : 0.0f;
}

// ---------------------------------------------------------------------------
// host launcher
// ---------------------------------------------------------------------------
extern "C" void kernel_launch(void* const* d_in, const int* in_sizes, int n_in,
                              void* d_out, int out_size, void* d_ws, size_t ws_size,
                              hipStream_t stream) {
    const float* x          = (const float*)d_in[0];
    const unsigned char* sm = (const unsigned char*)d_in[1];
    const float* pit_ctl    = (const float*)d_in[2];
    const float* ene_ctl    = (const float*)d_in[3];
    const float* dur_ctl    = (const float*)d_in[4];
    const float* pit_bins   = (const float*)d_in[5];
    const float* ene_bins   = (const float*)d_in[6];
    const float* pit_emb    = (const float*)d_in[7];
    const float* ene_emb    = (const float*)d_in[8];
    // predictor params: 9..18 dur, 19..28 pit, 29..38 ene; order w1,b1,g1,be1,w2,b2,g2,be2,wl,bl
    const float* P[3][10];
    for (int p = 0; p < 3; ++p)
        for (int j = 0; j < 10; ++j)
            P[p][j] = (const float*)d_in[9 + p * 10 + j];

    float* out = (float*)d_out;
    const long O_XEXP   = 0;
    const long O_PIT    = (long)BB * MAXMEL * HH;        // 16777216
    const long O_ENE    = O_PIT + (long)BB * SS;
    const long O_LOGDUR = O_ENE + (long)BB * SS;
    const long O_DUR    = O_LOGDUR + (long)BB * SS;
    const long O_MLEN   = O_DUR + (long)BB * SS;
    const long O_MMASK  = O_MLEN + BB;

    char* ws = (char*)d_ws;
    size_t o = 0;
    __bf16* xcolA = (__bf16*)(ws + o); o += (size_t)RT_N * KC_N * 512 * 2;  // 24 MB
    __bf16* hbuf  = (__bf16*)(ws + o); o += (size_t)BS * FF * 2;            // 8 MB
    __bf16* hcolA = (__bf16*)(ws + o); o += (size_t)RT_N * KC_N * 512 * 2;  // 24 MB
    float*  xbuf  = (float*)(ws + o);  o += (size_t)BS * HH * 4;            // 16 MB
    __bf16* w1f   = (__bf16*)(ws + o); o += (size_t)KC_N * CT_N * 512 * 2;  // 384 KB
    __bf16* w2f   = (__bf16*)(ws + o); o += (size_t)KC_N * CT_N * 512 * 2;  // 384 KB
    int*    cum   = (int*)(ws + o);    o += (size_t)BB * SS * 4;
    int*    mli   = (int*)(ws + o);    o += 128;
    (void)ws_size; (void)in_sizes; (void)n_in; (void)out_size;

    const int IM2COL_BLKS = (RT_N * KC_N * 512) / 256;   // 49152
    const int WFRAG_BLKS  = (KC_N * CT_N * 512) / 256;   // 768

    auto run_predictor = [&](int p, const __bf16* inA, float* outvec, const float* ctrl) {
        wfrag_kernel<<<WFRAG_BLKS, 256, 0, stream>>>(P[p][0], w1f);
        wfrag_kernel<<<WFRAG_BLKS, 256, 0, stream>>>(P[p][4], w2f);
        gemm_ln_kernel<0><<<RT_N, 128, 0, stream>>>(
            inA, w1f, P[p][1], P[p][2], P[p][3],
            hbuf, nullptr, nullptr, nullptr, nullptr, nullptr);
        im2col_frag_kernel<__bf16><<<IM2COL_BLKS, 256, 0, stream>>>(hbuf, hcolA);
        gemm_ln_kernel<1><<<RT_N, 128, 0, stream>>>(
            hcolA, w2f, P[p][5], P[p][6], P[p][7],
            nullptr, outvec, P[p][8], P[p][9], sm, ctrl);
    };

    // stage 1: duration + pitch predictors on original x
    im2col_frag_kernel<float><<<IM2COL_BLKS, 256, 0, stream>>>(x, xcolA);
    run_predictor(0, xcolA, out + O_LOGDUR, nullptr);
    run_predictor(1, xcolA, out + O_PIT, pit_ctl);

    // stage 2: x += pitch_emb[bucketize(pitch_pred)]; energy predictor
    add_emb_kernel<<<BS, HH, 0, stream>>>(out + O_PIT, pit_bins, pit_emb, x, xbuf);
    im2col_frag_kernel<float><<<IM2COL_BLKS, 256, 0, stream>>>(xbuf, xcolA);
    run_predictor(2, xcolA, out + O_ENE, ene_ctl);

    // stage 3: x += energy_emb[bucketize(energy_pred)] (in place in ws)
    add_emb_kernel<<<BS, HH, 0, stream>>>(out + O_ENE, ene_bins, ene_emb, xbuf, xbuf);

    // stage 4: durations, cumsum, length regulator
    duration_kernel<<<BB, SS, 0, stream>>>(out + O_LOGDUR, dur_ctl,
                                           out + O_DUR, cum, out + O_MLEN, mli);
    gather_kernel<<<BB * MAXMEL, HH, 0, stream>>>(xbuf, cum, mli,
                                                  out + O_XEXP, out + O_MMASK);
}
